// DiffBertSelfAttention_30142080483961
// MI455X (gfx1250) — matrix operand order repaired and checked by reading, not compiled, not run on hardware
//
#include <hip/hip_runtime.h>
#include <hip/hip_bf16.h>

#define B_   2
#define S_   2048
#define D_   1024
#define H_   16
#define DH   64
#define NQKV 3072   // 3*D (Q|K|V fused N dimension)

typedef __attribute__((ext_vector_type(16))) __bf16 v16bf;
typedef __attribute__((ext_vector_type(8)))  float  v8f;

union BF16Frag {
    v16bf v;
    unsigned int u[8];
};

#if __has_builtin(__builtin_amdgcn_global_load_async_to_lds_b128) && \
    __has_builtin(__builtin_amdgcn_s_wait_asynccnt)
#define HAVE_ASYNC_LDS 1
typedef int async_v4i __attribute__((vector_size(16)));
typedef __attribute__((address_space(1))) async_v4i* async_gptr;
typedef __attribute__((address_space(3))) async_v4i* async_lptr;
#define GL_ASYNC_B128(src, dst)                                   \
    __builtin_amdgcn_global_load_async_to_lds_b128(               \
        (async_gptr)(src), (async_lptr)(dst), 0, 0)
#else
#define HAVE_ASYNC_LDS 0
#endif

// fp32 -> bf16 via native converts (v_cvt on gfx1250), packed pair
__device__ __forceinline__ unsigned pack_bf16(float lo, float hi) {
    __bf16 l = (__bf16)lo;
    __bf16 h = (__bf16)hi;
    unsigned short ul, uh;
    __builtin_memcpy(&ul, &l, 2);
    __builtin_memcpy(&uh, &h, 2);
    return (unsigned)ul | ((unsigned)uh << 16);
}
__device__ __forceinline__ unsigned short f32_to_bf16(float f) {
    __bf16 b = (__bf16)f;
    unsigned short u;
    __builtin_memcpy(&u, &b, 2);
    return u;
}

// ---------------------------------------------------------------------------
// Kernel 1: fused QKV projection (hidden @ W^T) + RoPE on Q,K -> bf16 ws
// One wave computes one 16x16 tile of the (4096 x 3072) fused output.
// qkv layout: [which(3)][B][H][S][DH] bf16
// ---------------------------------------------------------------------------
__global__ __launch_bounds__(128) void qkv_rope_kernel(
    const float* __restrict__ hidden, const float* __restrict__ sinu,
    const float* __restrict__ Wq, const float* __restrict__ Wk,
    const float* __restrict__ Wv, unsigned short* __restrict__ qkv)
{
    const int lane = threadIdx.x & 31;
    const int wid  = threadIdx.x >> 5;
    const int hi   = lane >> 4;     // lane half (K sub-range select)
    const int nl   = lane & 15;

    const int NT   = NQKV / 16;     // 192 N-tiles
    int tile = blockIdx.x * 4 + wid;
    int mt = tile / NT;             // token tile (0..255)
    int ft = tile % NT;             // feature tile (0..191)
    int t0 = mt * 16;
    int f0 = ft * 16;
    int which = f0 / D_;            // 0=Q 1=K 2=V (uniform per wave)
    int fm    = f0 % D_;            // feature within matrix (multiple of 16)
    const float* W = (which == 0) ? Wq : ((which == 1) ? Wk : Wv);

    v8f c = {0.f, 0.f, 0.f, 0.f, 0.f, 0.f, 0.f, 0.f};

    const float* arow = hidden + (size_t)(t0 + nl) * D_;  // A row m = lane&15
    const float* wrow = W + (size_t)(fm + nl) * D_;       // B col n = lane&15

    for (int kk = 0; kk < D_; kk += 32) {
        BF16Frag a, bm;
#pragma unroll
        for (int i = 0; i < 8; ++i) {
            // A 16x32 bf16 layout: k = (i&3)*2 + (i>=4 ? 16:0) + hi*8
            int ka = ((i & 3) << 1) + ((i & 4) << 2) + (hi << 3);
            a.u[i] = pack_bf16(arow[kk + ka], arow[kk + ka + 1]);
        }
#pragma unroll
        for (int j = 0; j < 8; ++j) {
            // B 32x16 bf16 layout: k = (j&3)*2 + (j>=4 ? 16:0) + hi*8
            int kb = ((j & 3) << 1) + ((j & 4) << 2) + (hi << 3);
            bm.u[j] = pack_bf16(wrow[kk + kb], wrow[kk + kb + 1]);
        }
        c = __builtin_amdgcn_wmma_f32_16x16x32_bf16(
            false, a.v, false, bm.v, (short)0, c, false, false);
        __builtin_prefetch(arow + kk + 32, 0, 0);   // global_prefetch_b8
    }

    // Epilogue: C layout -> element (m,n): lane = n + 16*(m>>3), reg = m&7
    int head  = fm / DH;
    int dbase = fm & (DH - 1);
    int d     = dbase + nl;

    float vals[8];
#pragma unroll
    for (int r = 0; r < 8; ++r) vals[r] = c[r];

    if (which < 2) {   // RoPE on Q and K (uniform branch per wave)
        int j = d >> 1;
#pragma unroll
        for (int r = 0; r < 8; ++r) {
            int m = (hi << 3) + r;
            int s = (t0 + m) & (S_ - 1);
            float sv = sinu[s * DH + j];
            float cv = sinu[s * DH + 32 + j];
            float part = __shfl_xor(vals[r], 1, 32);   // partner feature d^1
            vals[r] = (d & 1) ? (vals[r] * cv + part * sv)
                              : (vals[r] * cv - part * sv);
        }
    }

#pragma unroll
    for (int r = 0; r < 8; ++r) {
        int m = (hi << 3) + r;
        int t = t0 + m;
        int b = t >> 11;            // t / S_
        int s = t & (S_ - 1);
        size_t off = ((size_t)which * B_ * H_ + (size_t)b * H_ + head) *
                         (size_t)(S_ * DH) +
                     (size_t)s * DH + d;
        qkv[off] = f32_to_bf16(vals[r]);
    }
}

// ---------------------------------------------------------------------------
// Kernel 2: flash attention. Block = 4 waves; each wave owns 16 query rows.
// Per 32-key tile: K/V staged into LDS (async-to-LDS on gfx1250), 4 WMMAs
// for Q*K^T, online softmax, P via LDS layout swap, 4 WMMAs for P*V.
// ---------------------------------------------------------------------------
__global__ __launch_bounds__(128) void attn_kernel(
    const unsigned short* __restrict__ qkv, const float* __restrict__ mask,
    float* __restrict__ out)
{
    __shared__ unsigned short pstage[4 * 16 * 32];   // per-wave 16x32 bf16
    __shared__ unsigned short ktile[32 * DH];        // 32 keys x 64 (4 KB)
    __shared__ unsigned short vtile[32 * DH];        // 32 keys x 64 (4 KB)

    const int lane = threadIdx.x & 31;
    const int wid  = threadIdx.x >> 5;
    const int hi   = lane >> 4;
    const int nl   = lane & 15;

    int bh = blockIdx.x >> 5;    // (b*H + h)
    int qt = blockIdx.x & 31;
    int b  = bh >> 4;
    int h  = bh & 15;

    const unsigned short* Q = qkv;
    const unsigned short* K = qkv + (size_t)B_ * H_ * S_ * DH;
    const unsigned short* V = K   + (size_t)B_ * H_ * S_ * DH;
    const size_t hb = (size_t)bh * S_ * DH;
    int q0 = qt * 64 + wid * 16;

    // Q A-fragments (16 rows x 64 K) -> two 16x32 fragments, resident
    BF16Frag qa0, qa1;
    {
        const unsigned short* qrow = Q + hb + (size_t)(q0 + nl) * DH;
#pragma unroll
        for (int i = 0; i < 8; ++i) {
            int ka = ((i & 3) << 1) + ((i & 4) << 2) + (hi << 3);
            qa0.u[i] = *(const unsigned int*)(qrow + ka);
            qa1.u[i] = *(const unsigned int*)(qrow + 32 + ka);
        }
    }

    float rmax[8], rsum[8];
#pragma unroll
    for (int r = 0; r < 8; ++r) { rmax[r] = -1e30f; rsum[r] = 0.f; }
    v8f acc[4];
#pragma unroll
    for (int ch = 0; ch < 4; ++ch)
        acc[ch] = (v8f){0.f, 0.f, 0.f, 0.f, 0.f, 0.f, 0.f, 0.f};

    unsigned short* myp = pstage + wid * 16 * 32;
    const int co = threadIdx.x * 8;   // 16-byte chunk per thread (ushort idx)

    for (int j0 = 0; j0 < S_; j0 += 32) {
        // ---- cooperative K/V tile staging: 2 x 4 KB contiguous regions ----
        const unsigned short* gk = K + hb + (size_t)j0 * DH;
        const unsigned short* gv = V + hb + (size_t)j0 * DH;
#if HAVE_ASYNC_LDS
        GL_ASYNC_B128(gk + co,        ktile + co);
        GL_ASYNC_B128(gk + co + 1024, ktile + co + 1024);
        GL_ASYNC_B128(gv + co,        vtile + co);
        GL_ASYNC_B128(gv + co + 1024, vtile + co + 1024);
        __builtin_amdgcn_s_wait_asynccnt(0);
#else
        *(uint4*)(ktile + co)        = *(const uint4*)(gk + co);
        *(uint4*)(ktile + co + 1024) = *(const uint4*)(gk + co + 1024);
        *(uint4*)(vtile + co)        = *(const uint4*)(gv + co);
        *(uint4*)(vtile + co + 1024) = *(const uint4*)(gv + co + 1024);
#endif
        __syncthreads();

        // ---- scores: Q (16x64) x K_tile^T (64x32) via 4 WMMAs ----
        v8f s0 = (v8f){0.f,0.f,0.f,0.f,0.f,0.f,0.f,0.f};
        v8f s1 = (v8f){0.f,0.f,0.f,0.f,0.f,0.f,0.f,0.f};
#pragma unroll
        for (int half = 0; half < 2; ++half) {
            BF16Frag kb0, kb1;
            const unsigned short* krow = ktile + (half * 16 + nl) * DH;
#pragma unroll
            for (int j = 0; j < 8; ++j) {
                int kb = ((j & 3) << 1) + ((j & 4) << 2) + (hi << 3);
                kb0.u[j] = *(const unsigned int*)(krow + kb);
                kb1.u[j] = *(const unsigned int*)(krow + 32 + kb);
            }
            v8f& sc = half ? s1 : s0;
            sc = __builtin_amdgcn_wmma_f32_16x16x32_bf16(
                false, qa0.v, false, kb0.v, (short)0, sc, false, false);
            sc = __builtin_amdgcn_wmma_f32_16x16x32_bf16(
                false, qa1.v, false, kb1.v, (short)0, sc, false, false);
        }

        // ---- online softmax over this 32-key tile ----
        float m0 = mask[b * S_ + j0 + nl];
        float m1 = mask[b * S_ + j0 + 16 + nl];
#pragma unroll
        for (int r = 0; r < 8; ++r) {
            float a0 = s0[r] * 0.125f + m0;   // 1/sqrt(64)
            float a1 = s1[r] * 0.125f + m1;
            float mx = fmaxf(a0, a1);
#pragma unroll
            for (int off = 1; off < 16; off <<= 1)
                mx = fmaxf(mx, __shfl_xor(mx, off, 32));
            float nm = fmaxf(rmax[r], mx);
            float sco = __expf(rmax[r] - nm);
            float p0 = __expf(a0 - nm);
            float p1 = __expf(a1 - nm);
            float ps = p0 + p1;
#pragma unroll
            for (int off = 1; off < 16; off <<= 1)
                ps += __shfl_xor(ps, off, 32);
            rsum[r] = rsum[r] * sco + ps;
            rmax[r] = nm;
#pragma unroll
            for (int ch = 0; ch < 4; ++ch) acc[ch][r] *= sco;
            int m = (hi << 3) + r;
            myp[m * 32 + nl]      = f32_to_bf16(p0);
            myp[m * 32 + 16 + nl] = f32_to_bf16(p1);
        }
        __syncthreads();

        // ---- P (16x32, A-layout from LDS) x V_tile (32x64) via 4 WMMAs ----
        BF16Frag pa;
#pragma unroll
        for (int i = 0; i < 8; ++i) {
            int ka = ((i & 3) << 1) + ((i & 4) << 2) + (hi << 3);
            pa.u[i] = *(const unsigned int*)(myp + nl * 32 + ka);
        }
#pragma unroll
        for (int ch = 0; ch < 4; ++ch) {
            BF16Frag vb;
#pragma unroll
            for (int j = 0; j < 8; ++j) {
                int kv = ((j & 3) << 1) + ((j & 4) << 2) + (hi << 3);   // key row within tile
                const unsigned short* vrow = vtile + kv * DH + ch * 16 + nl;
                vb.u[j] = (unsigned)vrow[0] | ((unsigned)vrow[DH] << 16);
            }
            acc[ch] = __builtin_amdgcn_wmma_f32_16x16x32_bf16(
                false, pa.v, false, vb.v, (short)0, acc[ch], false, false);
        }
        __syncthreads();
    }

    // ---- normalize and store fp32 ctx: out[b, s, h*64 + d] ----
#pragma unroll
    for (int ch = 0; ch < 4; ++ch) {
#pragma unroll
        for (int r = 0; r < 8; ++r) {
            int m = (hi << 3) + r;
            float val = acc[ch][r] / rsum[r];
            out[(size_t)(b * S_ + q0 + m) * D_ + h * DH + ch * 16 + nl] = val;
        }
    }
}

extern "C" void kernel_launch(void* const* d_in, const int* in_sizes, int n_in,
                              void* d_out, int out_size, void* d_ws, size_t ws_size,
                              hipStream_t stream) {
    const float* hidden = (const float*)d_in[0];
    const float* sinu   = (const float*)d_in[1];
    const float* mask   = (const float*)d_in[2];
    const float* Wq     = (const float*)d_in[3];
    const float* Wk     = (const float*)d_in[4];
    const float* Wv     = (const float*)d_in[5];
    unsigned short* qkv = (unsigned short*)d_ws;   // 3*B*H*S*DH bf16 = 24 MB
    float* out          = (float*)d_out;

    // 256 M-tiles * 192 N-tiles = 49152 wave-tiles / 4 waves per block
    qkv_rope_kernel<<<12288, 128, 0, stream>>>(hidden, sinu, Wq, Wk, Wv, qkv);
    // B*H*(S/64) = 2*16*32 = 1024 blocks
    attn_kernel<<<1024, 128, 0, stream>>>(qkv, mask, out);
}
